// UserEmbeddingAttentionLayer_18786186953286
// MI455X (gfx1250) — compile-verified, matrix operand order checked by reading, not run on hardware
//
#include <hip/hip_runtime.h>
#include <hip/hip_bf16.h>

// ---------------------------------------------------------------------------
// DIN attention, MI455X (gfx1250, wave32, WMMA).
//   feats@W1 folded:  h1 = sigmoid( c_q + ub @ Weff ),  Weff per-batch.
//   GEMM1: (208,128)x(128,80) f16 WMMA, GEMM2: (208,96)x(96,48) f16 WMMA.
//   Memory-bound (~420MB HBM @ 23.3TB/s ~ 18us); WMMA keeps compute far under.
// ---------------------------------------------------------------------------

typedef _Float16 v16h __attribute__((ext_vector_type(16)));
typedef _Float16 v8h  __attribute__((ext_vector_type(8)));
typedef float    v8f  __attribute__((ext_vector_type(8)));

#define NEG_LARGE (-4294967295.0f) // -2^32 + 1

// LDS layout (bytes)
#define OFF_A      0         // 208*128 f16  ub tile (f16), rows 200..207 zero
#define OFF_WEFF   53248     // 80*128  f16  Weff stored [h][f]
#define OFF_H1     73728     // 208*96  f16  hidden1 (cols 80..95 zero)
#define OFF_W2T    113664    // 48*96   f16  w2 transposed [n][k], zero padded
#define OFF_QS     122880    // 128 f32  query vector / phase-E partial sums
#define OFF_CFULL  123392    // 80  f32  per-b bias  q@(Wa+Wc)+b1
#define OFF_B2S    123712    // 48  f32  b2 padded
#define OFF_W3S    123904    // 48  f32  w3 padded
#define OFF_SCORES 124096    // 208 f32  attention logits / exp
#define OFF_RED    124928    // 2   f32  reduction scratch
#define SMEM_BYTES 124944

#define LP   208   // L padded to 13 m-tiles
#define L_   200
#define K1   128   // 2D
#define H1N  80
#define H1KP 96    // H1 padded for GEMM2 K
#define H2NP 48    // H2 padded to 3 n-tiles

// Fast sigmoid: v_exp_f32 + v_add + v_rcp_f32 (no IEEE div fixup chain).
__device__ __forceinline__ float sigmoidf_(float x) {
    return __builtin_amdgcn_rcpf(1.0f + __expf(-x));
}

// Load a 16x32 f16 A/B fragment (ISA 16-bit striping) from LDS, row-major,
// `stride` halves per row.
//   lane<16 : M=lane,     elems 0..7 at K=kb..kb+7, elems 8..15 at K=kb+16..kb+23
//   lane>=16: M=lane-16,  kb += 8
__device__ __forceinline__ v16h load_frag16(const _Float16* base, int row0,
                                            int stride, int kstep, int lane) {
    int r  = row0 + (lane & 15);
    int kb = kstep * 32 + ((lane >> 4) << 3);
    const _Float16* p = base + r * stride + kb;
    union { v16h v; v8h h[2]; } u;
    u.h[0] = *(const v8h*)(p);
    u.h[1] = *(const v8h*)(p + 16);
    return u.v;
}

__global__ void UserEmbeddingAttentionLayer_18786186953286_kernel(
    const float* __restrict__ queries,        // (B, 128)
    const float* __restrict__ user_behaviors, // (B, 200, 128)
    const int*   __restrict__ masks,          // (B, 200)
    const float* __restrict__ w1,             // (512, 80) row-major
    const float* __restrict__ b1,             // (80)
    const float* __restrict__ w2,             // (80, 40)
    const float* __restrict__ b2,             // (40)
    const float* __restrict__ w3,             // (40, 1)
    const float* __restrict__ b3,             // (1)
    float*       __restrict__ out)            // (B, 128)
{
    extern __shared__ char smem[];
    _Float16* A    = (_Float16*)(smem + OFF_A);
    _Float16* Weff = (_Float16*)(smem + OFF_WEFF);
    _Float16* H1   = (_Float16*)(smem + OFF_H1);
    _Float16* W2T  = (_Float16*)(smem + OFF_W2T);
    float*    qs     = (float*)(smem + OFF_QS);
    float*    cfull  = (float*)(smem + OFF_CFULL);
    float*    b2s    = (float*)(smem + OFF_B2S);
    float*    w3s    = (float*)(smem + OFF_W3S);
    float*    scores = (float*)(smem + OFF_SCORES);
    float*    red    = (float*)(smem + OFF_RED);

    const int b    = blockIdx.x;
    const int tid  = threadIdx.x;
    const int lane = tid & 31;
    const int wave = tid >> 5;

    const float* ubB = user_behaviors + (size_t)b * L_ * K1;

    // ---------------- Phase A: fills not depending on qs ----------------
    if (tid < K1) qs[tid] = queries[(size_t)b * K1 + tid];

    // zero H1 (pad cols matter for GEMM2 K)
    for (int i = tid; i < (LP * H1KP) / 2; i += 256) ((unsigned*)H1)[i] = 0u;
    // zero W2T (pads), then fill transposed w2
    for (int i = tid; i < (H2NP * H1KP) / 2; i += 256) ((unsigned*)W2T)[i] = 0u;
    // zero A pad rows 200..207
    for (int i = tid; i < (8 * K1) / 2; i += 256)
        ((unsigned*)(A + L_ * K1))[i] = 0u;
    // scores, small vectors
    for (int i = tid; i < LP; i += 256) scores[i] = 0.0f;
    if (tid < H2NP) {
        b2s[tid] = (tid < 40) ? b2[tid] : 0.0f;
        w3s[tid] = (tid < 40) ? w3[tid] : 0.0f;
    }
    if (tid >= 128 && tid < 128 + H1N) cfull[tid - 128] = b1[tid - 128];
    // ub -> f16 A tile (coalesced global read)
    for (int i = tid; i < L_ * K1; i += 256)
        A[i] = (_Float16)ubB[i];
    // w2 transpose fill: W2T[n][k] = w2[k][n]
    for (int i = tid; i < H1N * 40; i += 256) {
        int k = i / 40, n = i % 40;
        W2T[n * H1KP + k] = (_Float16)w2[k * 40 + n];
    }
    __syncthreads();

    // ---------------- Phase B: per-b weights (need qs) ----------------
    // Weff[h][f] = (w1[128+f][h] - w1[256+f][h]) + q[f]*w1[384+f][h]
    for (int i = tid; i < H1N * K1; i += 256) {
        int h = i >> 7, f = i & 127;
        float wu = w1[(size_t)(128 + f) * H1N + h] - w1[(size_t)(256 + f) * H1N + h];
        float wp = w1[(size_t)(384 + f) * H1N + h];
        Weff[h * K1 + f] = (_Float16)(wu + qs[f] * wp);
    }
    // cfull[h] += sum_f q[f]*(w1[f][h] + w1[256+f][h]) ; 240 threads, 3 f-segs
    if (tid < 240) {
        int h   = tid % H1N;
        int seg = tid / H1N;                 // 0..2
        int f0  = seg * 43;
        int f1  = (seg == 2) ? K1 : f0 + 43;
        float acc = 0.0f;
        for (int f = f0; f < f1; ++f)
            acc += qs[f] * (w1[(size_t)f * H1N + h] + w1[(size_t)(256 + f) * H1N + h]);
        atomicAdd(&cfull[h], acc);
    }
    // Warm the cache for the phase-E f32 re-read of ub (whole 128B lines).
    for (int i = tid; i < (L_ * K1) / 32; i += 256)
        __builtin_prefetch(ubB + i * 32, 0, 3);
    __syncthreads();

    // ---------------- Phase C: GEMM1  (208,128)x(128,80) ----------------
    // 13 m-tiles x 5 n-tiles = 65 jobs over 8 waves, 4 k-steps each.
    for (int job = wave; job < 65; job += 8) {
        int mt = job / 5, nt = job % 5;
        v8f c = {};
#pragma unroll
        for (int ks = 0; ks < 4; ++ks) {
            v16h a  = load_frag16(A,    mt * 16, K1, ks, lane);
            v16h bf = load_frag16(Weff, nt * 16, K1, ks, lane);
            c = __builtin_amdgcn_wmma_f32_16x16x32_f16(
                    false, a, false, bf, (short)0, c, false, false);
        }
        int n  = nt * 16 + (lane & 15);
        int m0 = mt * 16 + ((lane >> 4) << 3);
        float bias = cfull[n];
#pragma unroll
        for (int r = 0; r < 8; ++r)
            H1[(m0 + r) * H1KP + n] = (_Float16)sigmoidf_(c[r] + bias);
    }
    __syncthreads();

    // ---------------- Phase D: GEMM2 + fused score ----------------
    // (208,96)x(96,48): 13 x 3 = 39 jobs, 3 k-steps.
    for (int job = wave; job < 39; job += 8) {
        int mt = job / 3, nt = job % 3;
        v8f c = {};
#pragma unroll
        for (int ks = 0; ks < 3; ++ks) {
            v16h a  = load_frag16(H1,  mt * 16, H1KP, ks, lane);
            v16h bf = load_frag16(W2T, nt * 16, H1KP, ks, lane);
            c = __builtin_amdgcn_wmma_f32_16x16x32_f16(
                    false, a, false, bf, (short)0, c, false, false);
        }
        int n  = nt * 16 + (lane & 15);
        int m0 = mt * 16 + ((lane >> 4) << 3);
        float b2v = b2s[n], w3v = w3s[n];
#pragma unroll
        for (int r = 0; r < 8; ++r) {
            float v = sigmoidf_(c[r] + b2v) * w3v;
            // sum over the 16 columns of this row (lanes within 16-group)
            v += __shfl_xor(v, 1, 32);
            v += __shfl_xor(v, 2, 32);
            v += __shfl_xor(v, 4, 32);
            v += __shfl_xor(v, 8, 32);
            if ((lane & 15) == 0) atomicAdd(&scores[m0 + r], v);
        }
    }
    __syncthreads();

    // ---------------- Phase E: masked softmax + weighted sum ----------------
    if (tid < L_) {
        float s = scores[tid] + b3[0];
        if (masks[(size_t)b * L_ + tid] == 0) s = NEG_LARGE;
        scores[tid] = s;
    }
    __syncthreads();
    if (wave == 0) {
        float m = -3.4e38f;
        for (int l = lane; l < L_; l += 32) m = fmaxf(m, scores[l]);
        m = fmaxf(m, __shfl_xor(m, 16, 32));
        m = fmaxf(m, __shfl_xor(m,  8, 32));
        m = fmaxf(m, __shfl_xor(m,  4, 32));
        m = fmaxf(m, __shfl_xor(m,  2, 32));
        m = fmaxf(m, __shfl_xor(m,  1, 32));
        if (lane == 0) red[0] = m;
    }
    __syncthreads();
    float maxv = red[0];
    if (tid < L_) scores[tid] = __expf(scores[tid] - maxv);
    __syncthreads();
    if (wave == 0) {
        float s = 0.0f;
        for (int l = lane; l < L_; l += 32) s += scores[l];
        s += __shfl_xor(s, 16, 32);
        s += __shfl_xor(s,  8, 32);
        s += __shfl_xor(s,  4, 32);
        s += __shfl_xor(s,  2, 32);
        s += __shfl_xor(s,  1, 32);
        if (lane == 0) red[1] = s;
    }
    __syncthreads();
    // Weighted sum: all 256 threads. d = tid&127; half 0 does l=0..99,
    // half 1 does l=100..199 and parks its partial in qs[] (free by now).
    {
        int d  = tid & 127;
        int l0 = (tid < 128) ? 0 : 100;
        int l1 = (tid < 128) ? 100 : L_;
        const float* p = ubB + d;
        float acc = 0.0f;
        for (int l = l0; l < l1; ++l) acc += scores[l] * p[(size_t)l * K1];
        if (tid >= 128) qs[d] = acc;
        __syncthreads();
        if (tid < 128) {
            float inv = __builtin_amdgcn_rcpf(red[1]);
            out[(size_t)b * K1 + d] = (acc + qs[d]) * inv;
        }
    }
}

extern "C" void kernel_launch(void* const* d_in, const int* in_sizes, int n_in,
                              void* d_out, int out_size, void* d_ws, size_t ws_size,
                              hipStream_t stream) {
    const float* queries = (const float*)d_in[0];
    const float* ub      = (const float*)d_in[1];
    const int*   masks   = (const int*)d_in[2];
    const float* w1      = (const float*)d_in[3];
    const float* b1      = (const float*)d_in[4];
    const float* w2      = (const float*)d_in[5];
    const float* b2      = (const float*)d_in[6];
    const float* w3      = (const float*)d_in[7];
    const float* b3      = (const float*)d_in[8];
    float* out = (float*)d_out;

    const int B = 2048;
    UserEmbeddingAttentionLayer_18786186953286_kernel
        <<<B, 256, SMEM_BYTES, stream>>>(
            queries, ub, masks, w1, b1, w2, b2, w3, b3, out);
}
